// AttentionPooling_35115652612239
// MI455X (gfx1250) — matrix operand order repaired and checked by reading, not compiled, not run on hardware
//
#include <hip/hip_runtime.h>
#include <math.h>

typedef __attribute__((ext_vector_type(16))) _Float16 v16h;
typedef __attribute__((ext_vector_type(8)))  float    v8f;

#define HIDDEN 256
#define HHALF  128
#define RS     256   // LDS row stride (f16 elements) for W1^T

// ---------------------------------------------------------------------------
// Kernel 1: per-node attention score  s = tanh(x@W1 + b1) @ W2 + b2
// One wave32 computes 16 nodes; 8 waves / block -> 128 nodes / block.
// W1^T staged in 64KB LDS (f16), XOR-swizzled in 32B chunks for bank spread.
// 64 x v_wmma_f32_16x16x32_f16 per wave (8 K-steps x 8 N-tiles).
// ---------------------------------------------------------------------------
__global__ __launch_bounds__(256) void attn_score_wmma(
    const float* __restrict__ x, const float* __restrict__ W1,
    const float* __restrict__ b1, const float* __restrict__ W2,
    const float* __restrict__ b2, float* __restrict__ s, int N)
{
  __shared__ __align__(32) _Float16 w1t[HHALF * RS];   // 64 KB

  const int tid = threadIdx.x;
  // Stage W1^T (W1 is [256,128] row-major): w1t[n][k], swizzled at 16-elem
  // (32-byte) chunk granularity: chunk c of row n lives at chunk (c ^ (n&15)).
  for (int idx = tid; idx < HIDDEN * HHALF; idx += 256) {
    int k = idx >> 7;        // 0..255
    int n = idx & 127;       // 0..127
    int c = k >> 4, kin = k & 15;
    w1t[n * RS + (((c ^ (n & 15)) << 4) | kin)] = (_Float16)W1[idx];
  }
  __syncthreads();

  const int wave   = tid >> 5;
  const int lane   = tid & 31;
  const int laneLo = lane & 15;
  const int laneHi = lane >> 4;        // 0 or 1 (half-wave)

  const int nodeBase = (blockIdx.x * 8 + wave) * 16;
  int row = nodeBase + laneLo;
  if (row >= N) row = N - 1;           // clamp loads; stores guarded below
  const float* xrow = x + (size_t)row * HIDDEN;

  v8f acc[8] = {};

  #pragma unroll
  for (int kt = 0; kt < 8; ++kt) {
    // A fragment (contiguous K, matching B): half-wave 0 holds K 0..15,
    // half-wave 1 holds K 16..31 of this K-step.
    v16h a;
    const int k0 = kt * 32 + laneHi * 16;
    #pragma unroll
    for (int j = 0; j < 8; ++j) a[j]     = (_Float16)xrow[k0 + j];
    #pragma unroll
    for (int j = 0; j < 8; ++j) a[8 + j] = (_Float16)xrow[k0 + 8 + j];

    #pragma unroll
    for (int nt = 0; nt < 8; ++nt) {
      // B fragment (32x16 B layout): lane = column n, lanes 0-15 hold K 0-15,
      // lanes 16-31 hold K 16-31 -> one contiguous 32B chunk per lane.
      const int n = nt * 16 + laneLo;
      const int c = (kt * 2 + laneHi) ^ (n & 15);
      const v16h b = *(const v16h*)&w1t[n * RS + (c << 4)];
      acc[nt] = __builtin_amdgcn_wmma_f32_16x16x32_f16(
          false, a, false, b, (short)0, acc[nt], false, false);
    }
  }

  // Epilogue: +b1, tanh, dot with W2 along N, reduce across the 16 lanes of
  // each half-wave (C layout: VGPR r -> M=r (lanes 0-15) / M=r+8 (lanes 16-31),
  // N = lane&15).
  float p[8];
  #pragma unroll
  for (int r = 0; r < 8; ++r) p[r] = 0.0f;
  #pragma unroll
  for (int nt = 0; nt < 8; ++nt) {
    const int n = nt * 16 + laneLo;
    const float b1v = b1[n];
    const float w2v = W2[n];
    #pragma unroll
    for (int r = 0; r < 8; ++r)
      p[r] += tanhf(acc[nt][r] + b1v) * w2v;
  }
  #pragma unroll
  for (int off = 1; off < 16; off <<= 1) {
    #pragma unroll
    for (int r = 0; r < 8; ++r)
      p[r] += __shfl_xor(p[r], off, 32);
  }
  if (laneLo == 0) {
    const float b2v = b2[0];
    #pragma unroll
    for (int r = 0; r < 8; ++r) {
      const int node = nodeBase + laneHi * 8 + r;
      if (node < N) s[node] = p[r] + b2v;
    }
  }
}

// ---------------------------------------------------------------------------
// Kernel 2: per-graph node ranges via binary search on the sorted batch array.
// offsets[g] = lower_bound(batch, g), g in [0, B]; graph g owns
// [offsets[g], offsets[g+1]).
// ---------------------------------------------------------------------------
__global__ void graph_bounds(const int* __restrict__ batch, int N, int B,
                             int* __restrict__ offsets)
{
  const int g = blockIdx.x * blockDim.x + threadIdx.x;
  if (g > B) return;
  int lo = 0, hi = N;
  while (lo < hi) {
    const int mid = (lo + hi) >> 1;
    if (batch[mid] < g) lo = mid + 1; else hi = mid;
  }
  offsets[g] = lo;
}

// ---------------------------------------------------------------------------
// Kernel 3: one block per graph. Numerically-stable segmented softmax over s,
// then thread t accumulates hidden dim t of the weighted feature sum
// (coalesced 1KB row reads of x). Empty graphs write zeros.
// ---------------------------------------------------------------------------
__global__ __launch_bounds__(256) void segment_softmax_pool(
    const float* __restrict__ x, const float* __restrict__ s,
    const int* __restrict__ offsets, float* __restrict__ out)
{
  __shared__ float red[256];
  const int g   = blockIdx.x;
  const int tid = threadIdx.x;
  const int start = offsets[g];
  const int end   = offsets[g + 1];

  // segment max
  float m = -INFINITY;
  for (int i = start + tid; i < end; i += 256) m = fmaxf(m, s[i]);
  red[tid] = m; __syncthreads();
  for (int w = 128; w > 0; w >>= 1) {
    if (tid < w) red[tid] = fmaxf(red[tid], red[tid + w]);
    __syncthreads();
  }
  m = red[0]; __syncthreads();

  // segment sum of exp(s - m)
  float sm = 0.0f;
  for (int i = start + tid; i < end; i += 256) sm += __expf(s[i] - m);
  red[tid] = sm; __syncthreads();
  for (int w = 128; w > 0; w >>= 1) {
    if (tid < w) red[tid] += red[tid + w];
    __syncthreads();
  }
  const float inv = 1.0f / (red[0] + 1e-8f);

  // weighted feature reduction: thread t owns hidden dim t
  float acc = 0.0f;
  for (int i = start; i < end; ++i) {
    const float wv = __expf(s[i] - m) * inv;
    acc = fmaf(x[(size_t)i * HIDDEN + tid], wv, acc);
  }
  out[(size_t)g * HIDDEN + tid] = acc;
}

// ---------------------------------------------------------------------------
extern "C" void kernel_launch(void* const* d_in, const int* in_sizes, int n_in,
                              void* d_out, int out_size, void* d_ws, size_t ws_size,
                              hipStream_t stream)
{
  const float* x     = (const float*)d_in[0];
  const int*   batch = (const int*)d_in[1];   // JAX int64 downcasts to int32
  const float* W1    = (const float*)d_in[2];
  const float* b1    = (const float*)d_in[3];
  const float* W2    = (const float*)d_in[4];
  const float* b2    = (const float*)d_in[5];
  float* out = (float*)d_out;

  const int N = in_sizes[1];        // 262144 nodes
  const int B = out_size / HIDDEN;  // 1024 graphs

  float* s       = (float*)d_ws;
  int*   offsets = (int*)((char*)d_ws + (size_t)N * sizeof(float));

  attn_score_wmma<<<(N + 127) / 128, 256, 0, stream>>>(x, W1, b1, W2, b2, s, N);
  graph_bounds<<<(B + 256) / 256, 256, 0, stream>>>(batch, N, B, offsets);
  segment_softmax_pool<<<B, 256, 0, stream>>>(x, s, offsets, out);
}
